// VectorQuantizer_4260607558141
// MI455X (gfx1250) — matrix operand order repaired and checked by reading, not compiled, not run on hardware
//
#include <hip/hip_runtime.h>

#define BB 8
#define SS 2048
#define DD 256
#define KK 4096
#define NN (BB * SS)      // 16384 tokens
#define ND (NN * DD)      // 4194304 elements
#define NT (KK / 16)      // 256 codebook tiles

typedef __attribute__((ext_vector_type(16))) __bf16 v16bf;
typedef __attribute__((ext_vector_type(8)))  __bf16 v8bf;
typedef __attribute__((ext_vector_type(8)))  float  v8f;

// ---- async global->LDS copy (16B), ASYNCcnt-tracked ------------------------
// Builtin prototype (from hipcc diagnostic): (v4i as1*, v4i as3*, imm, imm)
typedef int b128i __attribute__((vector_size(16)));
typedef __attribute__((address_space(1))) b128i* as1_b128;
typedef __attribute__((address_space(3))) b128i* as3_b128;

#if __has_builtin(__builtin_amdgcn_global_load_async_to_lds_b128)
#define ASYNC_COPY16(g, l)                                                     \
    __builtin_amdgcn_global_load_async_to_lds_b128(                            \
        (as1_b128)(unsigned long long)(g),                                     \
        (as3_b128)(unsigned)(unsigned long long)(l), 0, 0)
#else
#define ASYNC_COPY16(g, l)                                                     \
    asm volatile("global_load_async_to_lds_b128 %0, %1, off" ::                \
                 "v"((unsigned)(unsigned long long)(l)),                       \
                 "v"((unsigned long long)(g)) : "memory")
#endif

#if __has_builtin(__builtin_amdgcn_s_wait_asynccnt)
#define WAIT_ASYNC(n) __builtin_amdgcn_s_wait_asynccnt(n)
#else
#define WAIT_ASYNC(n) asm volatile("s_wait_asynccnt %0" :: "i"(n) : "memory")
#endif

static __device__ __forceinline__ __bf16 f2bf(float f) {
    union { float f; unsigned u; } v; v.f = f;
    unsigned r = v.u + 0x7FFFu + ((v.u >> 16) & 1u);   // round-to-nearest-even
    unsigned short h = (unsigned short)(r >> 16);
    __bf16 o; __builtin_memcpy(&o, &h, 2);
    return o;
}

// ---------------------------------------------------------------- convert inputs
__global__ __launch_bounds__(256)
void vq_cvt_inputs(const float* __restrict__ in, __bf16* __restrict__ out) {
    int i = blockIdx.x * 256 + threadIdx.x;            // 8 elements per thread
    const float4* p = reinterpret_cast<const float4*>(in) + i * 2;
    float4 f0 = p[0], f1 = p[1];
    v8bf o = { f2bf(f0.x), f2bf(f0.y), f2bf(f0.z), f2bf(f0.w),
               f2bf(f1.x), f2bf(f1.y), f2bf(f1.z), f2bf(f1.w) };
    *(reinterpret_cast<v8bf*>(out) + i) = o;
}

// ------------------------------------------- convert codebook + per-code ||e||^2
__global__ __launch_bounds__(256)
void vq_cvt_codebook(const float* __restrict__ cbf, __bf16* __restrict__ cbb,
                     float* __restrict__ norms) {
    int wave = threadIdx.x >> 5, lane = threadIdx.x & 31;
    int k = blockIdx.x * 8 + wave;                     // one code row per wave
    const float4* src = reinterpret_cast<const float4*>(cbf + k * DD) + lane * 2;
    float4 f0 = src[0], f1 = src[1];
    v8bf o = { f2bf(f0.x), f2bf(f0.y), f2bf(f0.z), f2bf(f0.w),
               f2bf(f1.x), f2bf(f1.y), f2bf(f1.z), f2bf(f1.w) };
    *(reinterpret_cast<v8bf*>(cbb + k * DD) + lane) = o;
    float s = f0.x*f0.x + f0.y*f0.y + f0.z*f0.z + f0.w*f0.w
            + f1.x*f1.x + f1.y*f1.y + f1.z*f1.z + f1.w*f1.w;
    for (int m = 16; m; m >>= 1) s += __shfl_xor(s, m, 32);
    if (lane == 0) norms[k] = s;
}

// ---------------------------------------------------------------- zero accumulators
__global__ __launch_bounds__(256)
void vq_init(float* __restrict__ counts, float* __restrict__ loss_acc) {
    int i = blockIdx.x * 256 + threadIdx.x;
    if (i < KK) counts[i] = 0.0f;
    if (i == 0) loss_acc[0] = 0.0f;
}

// ------------------------------------------------- WMMA distance-GEMM + argmin
// Block = 128 threads (4 waves). Wave w owns an M-tile of 16 tokens; its A tile
// (16x256 bf16) stays in registers in the ISA A-layout. Codebook tiles (16 codes
// x 256 dims, 8KB) stream through a 4-deep LDS ring via GLOBAL_LOAD_ASYNC_TO_LDS,
// issued 3 tiles ahead and ordered with s_wait_asynccnt (async loads complete
// in order). Rows padded to 264 elems to spread LDS banks.
__global__ __launch_bounds__(128)
void vq_argmin(const __bf16* __restrict__ xb, const __bf16* __restrict__ cb,
               const float* __restrict__ norms, int* __restrict__ idx_out) {
    __shared__ __align__(16) __bf16 sB[4][16 * 264];   // 4 x 8448 B ring

    const int tid  = threadIdx.x;
    const int wave = tid >> 5;
    const int lane = tid & 31;
    const int n16  = lane & 15;
    const int h    = lane >> 4;
    const int m0   = blockIdx.x * 64 + wave * 16;
    const int r    = tid >> 3;                         // staging row 0..15
    const int seg  = tid & 7;                          // 64B segment within row

    // ---- preload A tile: lane holds row m0+n16; elems 0-7 = K h*8.., 8-15 = K 16+h*8..
    v16bf a[8];
    const __bf16* abase = xb + (m0 + n16) * DD;
    #pragma unroll
    for (int c = 0; c < 8; ++c) {
        int kd0 = c * 32;
        v8bf a0 = *reinterpret_cast<const v8bf*>(abase + kd0 + h * 8);
        v8bf a1 = *reinterpret_cast<const v8bf*>(abase + kd0 + 16 + h * 8);
        a[c] = __builtin_shufflevector(a0, a1, 0,1,2,3,4,5,6,7,8,9,10,11,12,13,14,15);
    }

    // async-stage one 16-code tile into ring slot `buf` (4 x b128 per thread)
    auto stage = [&](int tile, int buf) {
        const __bf16* g = cb + ((size_t)tile * 16 + r) * DD + seg * 32;
        __bf16*       l = &sB[buf][r * 264 + seg * 32];
        ASYNC_COPY16(g,      l);
        ASYNC_COPY16(g +  8, l +  8);
        ASYNC_COPY16(g + 16, l + 16);
        ASYNC_COPY16(g + 24, l + 24);
    };

    float bestVal[8];
    int   bestIdx[8];
    #pragma unroll
    for (int i = 0; i < 8; ++i) { bestVal[i] = 3.4e38f; bestIdx[i] = 0; }

    #pragma unroll
    for (int p = 0; p < 3; ++p) stage(p, p);           // prologue: 3 tiles in flight

    for (int t = 0; t < NT; ++t) {
        const int rem = NT - 1 - t;
        if (rem >= 3) { stage(t + 3, (t + 3) & 3); WAIT_ASYNC(12); }
        else if (rem == 2) WAIT_ASYNC(8);
        else if (rem == 1) WAIT_ASYNC(4);
        else               WAIT_ASYNC(0);
        if (t + 4 < NT)                                 // warm L2 ahead of the async engine
            __builtin_prefetch(cb + (size_t)(t + 4) * 16 * DD + tid * 32, 0, 1);
        __syncthreads();

        // ---- 8 WMMAs, two independent accumulator chains
        const float nr = norms[t * 16 + n16];
        const __bf16* bbase = &sB[t & 3][n16 * 264 + h * 8]; // B: lane = col, elems 0-7 = K h*8.., 8-15 = K 16+h*8..
        v8f acc0 = {0.f,0.f,0.f,0.f,0.f,0.f,0.f,0.f};
        v8f acc1 = {0.f,0.f,0.f,0.f,0.f,0.f,0.f,0.f};
        #pragma unroll
        for (int c = 0; c < 4; ++c) {
            v8bf b0 = *reinterpret_cast<const v8bf*>(bbase + (2 * c) * 32);
            v8bf b1 = *reinterpret_cast<const v8bf*>(bbase + (2 * c) * 32 + 16);
            v16bf bA = __builtin_shufflevector(b0, b1, 0,1,2,3,4,5,6,7,8,9,10,11,12,13,14,15);
            acc0 = __builtin_amdgcn_wmma_f32_16x16x32_bf16(
                       false, a[2 * c], false, bA, (short)0, acc0, false, false);
            v8bf b2 = *reinterpret_cast<const v8bf*>(bbase + (2 * c + 1) * 32);
            v8bf b3 = *reinterpret_cast<const v8bf*>(bbase + (2 * c + 1) * 32 + 16);
            v16bf bB = __builtin_shufflevector(b2, b3, 0,1,2,3,4,5,6,7,8,9,10,11,12,13,14,15);
            acc1 = __builtin_amdgcn_wmma_f32_16x16x32_bf16(
                       false, a[2 * c + 1], false, bB, (short)0, acc1, false, false);
        }

        // ---- score = ||e||^2 - 2 x.e ; running per-(row,col-slot) min
        const int code = t * 16 + n16;
        #pragma unroll
        for (int i = 0; i < 8; ++i) {
            float s = nr - 2.0f * acc0[i] - 2.0f * acc1[i]; // C layout: VGPR i -> row i+8h
            if (s < bestVal[i]) { bestVal[i] = s; bestIdx[i] = code; }
        }
        __syncthreads();   // all waves done with sB[t&3] before it is re-staged
    }

    // ---- argmin across the 16 columns (lanes within each 16-group)
    #pragma unroll
    for (int i = 0; i < 8; ++i) {
        float v = bestVal[i];
        int  ix = bestIdx[i];
        #pragma unroll
        for (int m = 8; m >= 1; m >>= 1) {
            float ov = __shfl_xor(v, m, 32);
            int   oi = __shfl_xor(ix, m, 32);
            if (ov < v || (ov == v && oi < ix)) { v = ov; ix = oi; }
        }
        if (n16 == 0) idx_out[m0 + i + 8 * h] = ix;
    }
}

// ------------------------------------- gather quantized, loss partial, counts, idx
__global__ __launch_bounds__(256)
void vq_gather_loss(const float* __restrict__ inputs, const int* __restrict__ mask,
                    const float* __restrict__ cbf, const int* __restrict__ idx,
                    float* __restrict__ out_q, float* __restrict__ out_idx,
                    float* __restrict__ counts, float* __restrict__ loss_acc) {
    const int token = blockIdx.x;
    const int d     = threadIdx.x;
    const int code  = idx[token];
    const float m   = (float)mask[token];
    float q = cbf[code * DD + d] * m;                  // masked quantized
    float x = inputs[token * DD + d];
    out_q[token * DD + d] = q;                         // quantized_st value == quantized
    float diff = q - x * m;
    float sq = diff * diff;
    for (int o = 16; o; o >>= 1) sq += __shfl_xor(sq, o, 32);
    __shared__ float part[8];
    if ((threadIdx.x & 31) == 0) part[threadIdx.x >> 5] = sq;
    __syncthreads();
    if (threadIdx.x == 0) {
        float s = 0.f;
        #pragma unroll
        for (int w = 0; w < 8; ++w) s += part[w];
        atomicAdd(loss_acc, s);
        atomicAdd(&counts[code], m);
        out_idx[token] = (m != 0.f) ? (float)code : 0.f;
    }
}

// ---------------------------------------------------------- loss & perplexity scalars
__global__ __launch_bounds__(256)
void vq_finalize(const float* __restrict__ counts, const float* __restrict__ loss_acc,
                 const int* __restrict__ mask,
                 float* __restrict__ out_loss, float* __restrict__ out_ppl) {
    __shared__ float sh[8];
    float nv = 0.f;
    for (int i = threadIdx.x; i < NN; i += 256) nv += (float)mask[i];
    for (int o = 16; o; o >>= 1) nv += __shfl_xor(nv, o, 32);
    if ((threadIdx.x & 31) == 0) sh[threadIdx.x >> 5] = nv;
    __syncthreads();
    float numValid = 0.f;
    #pragma unroll
    for (int w = 0; w < 8; ++w) numValid += sh[w];
    __syncthreads();
    float hacc = 0.f;
    for (int k = threadIdx.x; k < KK; k += 256) {
        float p = counts[k] / numValid;
        hacc += p * logf(p + 1e-10f);
    }
    for (int o = 16; o; o >>= 1) hacc += __shfl_xor(hacc, o, 32);
    if ((threadIdx.x & 31) == 0) sh[threadIdx.x >> 5] = hacc;
    __syncthreads();
    if (threadIdx.x == 0) {
        float H = 0.f;
        #pragma unroll
        for (int w = 0; w < 8; ++w) H += sh[w];
        out_loss[0] = 1.25f * loss_acc[0] / numValid;  // q_loss + 0.25*e_loss (same fwd value)
        out_ppl[0]  = expf(-H);
    }
}

extern "C" void kernel_launch(void* const* d_in, const int* in_sizes, int n_in,
                              void* d_out, int out_size, void* d_ws, size_t ws_size,
                              hipStream_t stream) {
    (void)in_sizes; (void)n_in; (void)out_size; (void)ws_size;
    const float* inputs = (const float*)d_in[0];   // [N, D] f32
    const int*   mask   = (const int*)d_in[1];     // [N] i32
    const float* cbf    = (const float*)d_in[2];   // [K, D] f32
    float* out = (float*)d_out;                    // [N*D | loss | ppl | N idx]

    char* ws = (char*)d_ws;
    __bf16* xb      = (__bf16*)(ws);                                   // 8 MB
    __bf16* cbb     = (__bf16*)(ws + (size_t)ND * 2);                  // 2 MB
    float*  norms   = (float*) (ws + (size_t)ND * 2 + (size_t)KK * DD * 2);
    int*    idx_ws  = (int*)   ((char*)norms + (size_t)KK * 4);
    float*  counts  = (float*) ((char*)idx_ws + (size_t)NN * 4);
    float*  lossacc = (float*) ((char*)counts + (size_t)KK * 4);

    vq_cvt_inputs  <<<ND / (256 * 8), 256, 0, stream>>>(inputs, xb);
    vq_cvt_codebook<<<KK / 8,         256, 0, stream>>>(cbf, cbb, norms);
    vq_init        <<<KK / 256,       256, 0, stream>>>(counts, lossacc);
    vq_argmin      <<<NN / 64,        128, 0, stream>>>(xb, cbb, norms, idx_ws);
    vq_gather_loss <<<NN,             256, 0, stream>>>(inputs, mask, cbf, idx_ws,
                                                        out, out + ND + 2, counts, lossacc);
    vq_finalize    <<<1,              256, 0, stream>>>(counts, lossacc, mask,
                                                        out + ND, out + ND + 1);
}
